// SimpleMoE_17093969838179
// MI455X (gfx1250) — matrix operand order repaired and checked by reading, not compiled, not run on hardware
//
#include <hip/hip_runtime.h>

#define HIDDEN 1024
#define FFN    4096
#define NEXP   8
#define NTOK   16384

#define BM 64
#define BN 128
#define BK 32
#define LDK 40   // padded K-stride (elements) for LDS tiles

typedef __attribute__((ext_vector_type(16))) _Float16 v16h;
typedef __attribute__((ext_vector_type(8)))  _Float16 v8h;
typedef __attribute__((ext_vector_type(4)))  _Float16 v4h;
typedef __attribute__((ext_vector_type(8)))  float    v8f;

// ---- WMMA fragment loaders from LDS (row-major [16][LDK] tiles) ----
// A (16x32 f16): lane<16 holds row=lane, K={0..7,16..23}; lane>=16 row=lane-16, K={8..15,24..31}
__device__ __forceinline__ v16h load_a_frag(const _Float16* base, int lane) {
  const int r  = lane & 15;
  const int kh = (lane >> 4) << 3;           // 0 or 8
  const _Float16* p = base + r * LDK + kh;
  v8h lo = *(const v8h*)p;                   // K = kh .. kh+7
  v8h hi = *(const v8h*)(p + 16);            // K = 16+kh .. 16+kh+7
  v16h out;
#pragma unroll
  for (int i = 0; i < 8; ++i) { out[i] = lo[i]; out[i + 8] = hi[i]; }
  return out;
}

// B (32x16 f16), staged as [n][k]: lane<16 holds col=lane, K={0..7,16..23}; lane>=16 col=lane-16, K={8..15,24..31}
__device__ __forceinline__ v16h load_b_frag(const _Float16* base, int lane) {
  const int c  = lane & 15;
  const int kh = (lane >> 4) << 3;           // 0 or 8
  const _Float16* p = base + c * LDK + kh;
  v8h lo = *(const v8h*)p;
  v8h hi = *(const v8h*)(p + 16);
  v16h out;
#pragma unroll
  for (int i = 0; i < 8; ++i) { out[i] = lo[i]; out[i + 8] = hi[i]; }
  return out;
}

// ---- fp32 -> f16 elementwise (x) ----
__global__ __launch_bounds__(256) void convert_f16(const float* __restrict__ in,
                                                   _Float16* __restrict__ out, size_t n) {
  size_t i = ((size_t)blockIdx.x * 256 + threadIdx.x) * 4;
  if (i >= n) return;
  float4 v = *(const float4*)(in + i);
  v4h o = { (_Float16)v.x, (_Float16)v.y, (_Float16)v.z, (_Float16)v.w };
  *(v4h*)(out + i) = o;
}

// ---- fp32 [R][C] -> f16 [C][R], batched over blockIdx.z ----
__global__ __launch_bounds__(256) void transpose_convert(const float* __restrict__ in,
                                                         _Float16* __restrict__ out,
                                                         int R, int C) {
  __shared__ float tile[32][33];
  const size_t base = (size_t)blockIdx.z * R * C;
  const int c0 = blockIdx.x * 32, r0 = blockIdx.y * 32;
  const int tx = threadIdx.x & 31, ty = threadIdx.x >> 5;
#pragma unroll
  for (int j = 0; j < 4; ++j)
    tile[ty + j * 8][tx] = in[base + (size_t)(r0 + ty + j * 8) * C + c0 + tx];
  __syncthreads();
#pragma unroll
  for (int j = 0; j < 4; ++j)
    out[base + (size_t)(c0 + ty + j * 8) * R + r0 + tx] = (_Float16)tile[tx][ty + j * 8];
}

// ---- gating: scores, top-2, softmax over 2, atomic compaction into per-expert lists ----
__global__ __launch_bounds__(256) void gate_topk(const float* __restrict__ x,
                                                 const float* __restrict__ Wg,
                                                 const float* __restrict__ bg,
                                                 int* __restrict__ counts,
                                                 int* __restrict__ list,
                                                 float* __restrict__ wgt) {
  __shared__ float sW[HIDDEN * NEXP];   // 32 KB
  for (int i = threadIdx.x; i < HIDDEN * NEXP; i += 256) sW[i] = Wg[i];
  __syncthreads();
  const int t = blockIdx.x * 256 + threadIdx.x;
  float s[NEXP];
#pragma unroll
  for (int j = 0; j < NEXP; ++j) s[j] = bg[j];
  const float4* xr = (const float4*)(x + (size_t)t * HIDDEN);
  for (int q = 0; q < HIDDEN / 4; ++q) {
    float4 v = xr[q];
    const float* w = &sW[q * 32];
#pragma unroll
    for (int j = 0; j < NEXP; ++j)
      s[j] += v.x * w[j] + v.y * w[8 + j] + v.z * w[16 + j] + v.w * w[24 + j];
  }
  float b0 = s[0], b1v = -3.4e38f; int i0 = 0, i1 = 1;
#pragma unroll
  for (int j = 1; j < NEXP; ++j) {
    if (s[j] > b0)       { b1v = b0; i1 = i0; b0 = s[j]; i0 = j; }
    else if (s[j] > b1v) { b1v = s[j]; i1 = j; }
  }
  const float e1 = __expf(b1v - b0);
  const float inv = 1.0f / (1.0f + e1);
  const float w0 = inv, w1 = e1 * inv;
  int p0 = atomicAdd(&counts[i0], 1);
  list[i0 * NTOK + p0] = t; wgt[i0 * NTOK + p0] = w0;
  int p1 = atomicAdd(&counts[i1], 1);
  list[i1 * NTOK + p1] = t; wgt[i1 * NTOK + p1] = w1;
}

// ---- GEMM1: H = relu(gather(xh) @ W1^T + b1), f16 out ----
__global__ __launch_bounds__(256) void moe_gemm1(const _Float16* __restrict__ xh,   // [NTOK][HIDDEN]
                                                 const _Float16* __restrict__ w1t,  // [FFN][HIDDEN]
                                                 const float* __restrict__ b1,      // [FFN]
                                                 const int* __restrict__ list,
                                                 const int* __restrict__ countp,
                                                 _Float16* __restrict__ Hbuf) {     // [NTOK][FFN]
  const int cnt = *countp;
  const int m0 = blockIdx.y * BM;
  if (m0 >= cnt) return;
  const int n0 = blockIdx.x * BN;

  __shared__ _Float16 As[2][BM * LDK];
  __shared__ _Float16 Bs[2][BN * LDK];

  const int tid = threadIdx.x, lane = tid & 31, wave = tid >> 5;
  // A staging: 4 threads/row, 16B each (gathered rows)
  const int arow = tid >> 2, aseg = (tid & 3) * 8;
  int am = m0 + arow; if (am >= cnt) am = cnt - 1;         // clamp padding rows
  const _Float16* aptr = xh + (size_t)list[am] * HIDDEN + aseg;
  // B staging: 2 threads/row(n), 32B each
  const int brow = tid >> 1, bseg = (tid & 1) * 16;
  const _Float16* bptr = w1t + (size_t)(n0 + brow) * HIDDEN + bseg;

  // 2x2 tile mapping per wave: 4 row-tiles x 8 col-tiles over 8 waves
  const int rtb = (wave >> 2) * 2;   // 0 or 2
  const int ctb = (wave & 3) * 2;    // 0,2,4,6
  v8f acc00 = {}, acc01 = {}, acc10 = {}, acc11 = {};

  // preload chunk 0
  *(v8h*)&As[0][arow * LDK + aseg]     = *(const v8h*)(aptr);
  *(v8h*)&Bs[0][brow * LDK + bseg]     = *(const v8h*)(bptr);
  *(v8h*)&Bs[0][brow * LDK + bseg + 8] = *(const v8h*)(bptr + 8);

  int buf = 0;
  for (int kc = 0; kc < HIDDEN; kc += BK) {
    const bool more = (kc + BK) < HIDDEN;
    v8h ga = {}, gb0 = {}, gb1 = {};
    if (more) {
      ga  = *(const v8h*)(aptr + kc + BK);
      gb0 = *(const v8h*)(bptr + kc + BK);
      gb1 = *(const v8h*)(bptr + kc + BK + 8);
    }
    __syncthreads();  // stores into As/Bs[buf] complete
    v16h a0 = load_a_frag(&As[buf][(rtb + 0) * 16 * LDK], lane);
    v16h a1 = load_a_frag(&As[buf][(rtb + 1) * 16 * LDK], lane);
    v16h b0 = load_b_frag(&Bs[buf][(ctb + 0) * 16 * LDK], lane);
    v16h b1 = load_b_frag(&Bs[buf][(ctb + 1) * 16 * LDK], lane);
    acc00 = __builtin_amdgcn_wmma_f32_16x16x32_f16(false, a0, false, b0, (short)0, acc00, false, false);
    acc01 = __builtin_amdgcn_wmma_f32_16x16x32_f16(false, a0, false, b1, (short)0, acc01, false, false);
    acc10 = __builtin_amdgcn_wmma_f32_16x16x32_f16(false, a1, false, b0, (short)0, acc10, false, false);
    acc11 = __builtin_amdgcn_wmma_f32_16x16x32_f16(false, a1, false, b1, (short)0, acc11, false, false);
    if (more) {
      const int nb = buf ^ 1;
      *(v8h*)&As[nb][arow * LDK + aseg]     = ga;
      *(v8h*)&Bs[nb][brow * LDK + bseg]     = gb0;
      *(v8h*)&Bs[nb][brow * LDK + bseg + 8] = gb1;
    }
    buf ^= 1;
  }

  const int col = lane & 15, rbase = (lane < 16) ? 0 : 8;
  v8f accs[2][2] = {{acc00, acc01}, {acc10, acc11}};
#pragma unroll
  for (int r = 0; r < 2; ++r) {
#pragma unroll
    for (int c = 0; c < 2; ++c) {
      const int f = n0 + (ctb + c) * 16 + col;
      const float bias = b1[f];
#pragma unroll
      for (int i = 0; i < 8; ++i) {
        const int m = m0 + (rtb + r) * 16 + rbase + i;
        float v = accs[r][c][i] + bias;
        v = v > 0.f ? v : 0.f;
        Hbuf[(size_t)m * FFN + f] = (_Float16)v;
      }
    }
  }
}

// ---- GEMM2: out[token] += cw * (H @ W2^T + b2) ----
__global__ __launch_bounds__(256) void moe_gemm2(const _Float16* __restrict__ Hbuf, // [NTOK][FFN]
                                                 const _Float16* __restrict__ w2t,  // [HIDDEN][FFN]
                                                 const float* __restrict__ b2,      // [HIDDEN]
                                                 const int* __restrict__ list,
                                                 const float* __restrict__ wgt,
                                                 const int* __restrict__ countp,
                                                 float* __restrict__ out) {         // [NTOK][HIDDEN]
  const int cnt = *countp;
  const int m0 = blockIdx.y * BM;
  if (m0 >= cnt) return;
  const int n0 = blockIdx.x * BN;

  __shared__ _Float16 As[2][BM * LDK];
  __shared__ _Float16 Bs[2][BN * LDK];

  const int tid = threadIdx.x, lane = tid & 31, wave = tid >> 5;
  const int arow = tid >> 2, aseg = (tid & 3) * 8;
  const _Float16* aptr = Hbuf + (size_t)(m0 + arow) * FFN + aseg;  // padded rows written by gemm1
  const int brow = tid >> 1, bseg = (tid & 1) * 16;
  const _Float16* bptr = w2t + (size_t)(n0 + brow) * FFN + bseg;

  const int rtb = (wave >> 2) * 2;
  const int ctb = (wave & 3) * 2;
  v8f acc00 = {}, acc01 = {}, acc10 = {}, acc11 = {};

  *(v8h*)&As[0][arow * LDK + aseg]     = *(const v8h*)(aptr);
  *(v8h*)&Bs[0][brow * LDK + bseg]     = *(const v8h*)(bptr);
  *(v8h*)&Bs[0][brow * LDK + bseg + 8] = *(const v8h*)(bptr + 8);

  int buf = 0;
  for (int kc = 0; kc < FFN; kc += BK) {
    const bool more = (kc + BK) < FFN;
    v8h ga = {}, gb0 = {}, gb1 = {};
    if (more) {
      ga  = *(const v8h*)(aptr + kc + BK);
      gb0 = *(const v8h*)(bptr + kc + BK);
      gb1 = *(const v8h*)(bptr + kc + BK + 8);
    }
    __syncthreads();
    v16h a0 = load_a_frag(&As[buf][(rtb + 0) * 16 * LDK], lane);
    v16h a1 = load_a_frag(&As[buf][(rtb + 1) * 16 * LDK], lane);
    v16h b0 = load_b_frag(&Bs[buf][(ctb + 0) * 16 * LDK], lane);
    v16h b1 = load_b_frag(&Bs[buf][(ctb + 1) * 16 * LDK], lane);
    acc00 = __builtin_amdgcn_wmma_f32_16x16x32_f16(false, a0, false, b0, (short)0, acc00, false, false);
    acc01 = __builtin_amdgcn_wmma_f32_16x16x32_f16(false, a0, false, b1, (short)0, acc01, false, false);
    acc10 = __builtin_amdgcn_wmma_f32_16x16x32_f16(false, a1, false, b0, (short)0, acc10, false, false);
    acc11 = __builtin_amdgcn_wmma_f32_16x16x32_f16(false, a1, false, b1, (short)0, acc11, false, false);
    if (more) {
      const int nb = buf ^ 1;
      *(v8h*)&As[nb][arow * LDK + aseg]     = ga;
      *(v8h*)&Bs[nb][brow * LDK + bseg]     = gb0;
      *(v8h*)&Bs[nb][brow * LDK + bseg + 8] = gb1;
    }
    buf ^= 1;
  }

  const int col = lane & 15, rbase = (lane < 16) ? 0 : 8;
  v8f accs[2][2] = {{acc00, acc01}, {acc10, acc11}};
  int tok[2][8]; float wv[2][8]; bool valid[2][8];
#pragma unroll
  for (int r = 0; r < 2; ++r) {
#pragma unroll
    for (int i = 0; i < 8; ++i) {
      const int m = m0 + (rtb + r) * 16 + rbase + i;
      valid[r][i] = (m < cnt);
      const int ms = valid[r][i] ? m : (cnt - 1);
      tok[r][i] = list[ms];
      wv[r][i]  = wgt[ms];
    }
  }
#pragma unroll
  for (int r = 0; r < 2; ++r) {
#pragma unroll
    for (int c = 0; c < 2; ++c) {
      const int n = n0 + (ctb + c) * 16 + col;
      const float bias = b2[n];
#pragma unroll
      for (int i = 0; i < 8; ++i) {
        if (valid[r][i]) {
          float* o = out + (size_t)tok[r][i] * HIDDEN + n;
          *o += wv[r][i] * (accs[r][c][i] + bias);   // race-free: unique (token, n) per launch
        }
      }
    }
  }
}

extern "C" void kernel_launch(void* const* d_in, const int* in_sizes, int n_in,
                              void* d_out, int out_size, void* d_ws, size_t ws_size,
                              hipStream_t stream) {
  (void)in_sizes; (void)n_in; (void)ws_size;
  const float* x  = (const float*)d_in[0];
  const float* Wg = (const float*)d_in[1];
  const float* bg = (const float*)d_in[2];
  const float* W1 = (const float*)d_in[3];
  const float* b1 = (const float*)d_in[4];
  const float* W2 = (const float*)d_in[5];
  const float* b2 = (const float*)d_in[6];
  float* out = (float*)d_out;

  char* ws = (char*)d_ws;
  size_t off = 0;
  auto take = [&](size_t bytes) -> void* {
    void* p = (void*)(ws + off);
    off += (bytes + 255) & ~(size_t)255;
    return p;
  };
  int*      counts = (int*)take(NEXP * sizeof(int));
  int*      list   = (int*)take((size_t)NEXP * NTOK * sizeof(int));
  float*    wgt    = (float*)take((size_t)NEXP * NTOK * sizeof(float));
  _Float16* xh     = (_Float16*)take((size_t)NTOK * HIDDEN * 2);
  _Float16* w1t    = (_Float16*)take((size_t)NEXP * FFN * HIDDEN * 2);
  _Float16* w2t    = (_Float16*)take((size_t)NEXP * HIDDEN * FFN * 2);
  _Float16* Hbuf   = (_Float16*)take((size_t)NTOK * FFN * 2);

  hipMemsetAsync(counts, 0, NEXP * sizeof(int), stream);
  hipMemsetAsync(d_out, 0, (size_t)out_size * sizeof(float), stream);

  convert_f16<<<(NTOK * HIDDEN) / (256 * 4), 256, 0, stream>>>(x, xh, (size_t)NTOK * HIDDEN);
  transpose_convert<<<dim3(FFN / 32, HIDDEN / 32, NEXP), 256, 0, stream>>>(W1, w1t, HIDDEN, FFN);
  transpose_convert<<<dim3(HIDDEN / 32, FFN / 32, NEXP), 256, 0, stream>>>(W2, w2t, FFN, HIDDEN);
  gate_topk<<<NTOK / 256, 256, 0, stream>>>(x, Wg, bg, counts, list, wgt);

  for (int e = 0; e < NEXP; ++e) {
    moe_gemm1<<<dim3(FFN / BN, NTOK / BM), 256, 0, stream>>>(
        xh, w1t + (size_t)e * FFN * HIDDEN, b1 + (size_t)e * FFN,
        list + e * NTOK, counts + e, Hbuf);
    moe_gemm2<<<dim3(HIDDEN / BN, NTOK / BM), 256, 0, stream>>>(
        Hbuf, w2t + (size_t)e * HIDDEN * FFN, b2 + (size_t)e * HIDDEN,
        list + e * NTOK, wgt + e * NTOK, counts + e, out);
  }
}
